// RSNU_60739427500523
// MI455X (gfx1250) — matrix operand order repaired and checked by reading, not compiled, not run on hardware
//
#include <hip/hip_runtime.h>

#define L_TAU 0.8f

typedef __attribute__((ext_vector_type(8)))  __bf16 v8bf;
typedef __attribute__((ext_vector_type(16))) __bf16 v16bf;
typedef __attribute__((ext_vector_type(8)))  float  v8f;
typedef __attribute__((ext_vector_type(4)))  float  v4f;

__device__ __forceinline__ unsigned short f32_to_bf16_bits(float f) {
  unsigned int u = __float_as_uint(f);
  u = (u + 0x7FFFu + ((u >> 16) & 1u)) >> 16;   // round-to-nearest-even
  return (unsigned short)u;
}

__device__ __forceinline__ __bf16 bits_to_bf16(unsigned short h) {
  __bf16 r;
  __builtin_memcpy(&r, &h, 2);
  return r;
}

// A-fragment (16x32 bf16): within a row-contiguous 32-K chunk, lane holds
// row M=lane&15, 16 consecutive K starting at (lane>>4)*16
__device__ __forceinline__ v16bf load_a_frag(const unsigned short* p) {
  v8bf lo = *(const v8bf*)(const void*)(p);
  v8bf hi = *(const v8bf*)(const void*)(p + 8);
  v16bf a;
#pragma unroll
  for (int i = 0; i < 8; ++i) { a[i] = lo[i]; a[i + 8] = hi[i]; }
  return a;
}

// B-fragment from f32 weights: lane holds col N=lane&15, 16 consecutive K
__device__ __forceinline__ v16bf load_b_frag_f32(const float* p) {
  const v4f* p4 = (const v4f*)(const void*)(p);
  v16bf b;
#pragma unroll
  for (int q = 0; q < 4; ++q) {
    v4f w = p4[q];
#pragma unroll
    for (int j = 0; j < 4; ++j) b[q * 4 + j] = bits_to_bf16(f32_to_bf16_bits(w[j]));
  }
  return b;
}

// B-fragment from bf16 weights (16 consecutive K bf16, no convert)
__device__ __forceinline__ v16bf load_w_frag_bf16(const unsigned short* p) {
  v8bf lo = *(const v8bf*)(const void*)(p);
  v8bf hi = *(const v8bf*)(const void*)(p + 8);
  v16bf b;
#pragma unroll
  for (int i = 0; i < 8; ++i) { b[i] = lo[i]; b[i + 8] = hi[i]; }
  return b;
}

// ---------------------------------------------------------------------------
__global__ void k_zero_u32(unsigned int* __restrict__ p, int n) {
  int i = blockIdx.x * blockDim.x + threadIdx.x;
  if (i < n) p[i] = 0u;
}

__global__ void k_cvt_bf16(const float* __restrict__ src,
                           unsigned short* __restrict__ dst, int n) {
  int i = blockIdx.x * blockDim.x + threadIdx.x;
  if (i < n) dst[i] = f32_to_bf16_bits(src[i]);
}

// x (B,2,T,128,128) f32 -> Xb (T*B, 32768) bf16 row-major (fallback path)
__global__ void k_pack_x_bf16(const float* __restrict__ x,
                              unsigned short* __restrict__ Xb, int total) {
  int i = blockIdx.x * blockDim.x + threadIdx.x;
  if (i >= total) return;
  int c  = i & 32767;
  int tb = i >> 15;
  int t  = tb >> 4;
  int b  = tb & 15;
  int c1 = c >> 14;
  int hw = c & 16383;
  size_t src = ((size_t)((b * 2 + c1) * 20 + t) << 14) + (size_t)hw;
  Xb[i] = f32_to_bf16_bits(x[src]);
}

// x -> K-blocked layout Xb2[K/32][320][32] bf16 (full path); all 20 M-tile
// fragments of one 32-K chunk live within 20KB of one base pointer.
__global__ void k_pack_x_kblk(const float* __restrict__ x,
                              unsigned short* __restrict__ Xb, int total) {
  int i = blockIdx.x * blockDim.x + threadIdx.x;
  if (i >= total) return;
  int c   = i & 32767;
  int row = i >> 15;           // t*16 + b
  int t   = row >> 4;
  int b   = row & 15;
  int c1  = c >> 14;
  int hw  = c & 16383;
  size_t src = ((size_t)((b * 2 + c1) * 20 + t) << 14) + (size_t)hw;
  size_t dst = ((size_t)(c >> 5) * 320 + row) * 32 + (c & 31);
  Xb[dst] = f32_to_bf16_bits(x[src]);
}

// ---------------------------------------------------------------------------
// Big GEMM, W1 streamed from HBM exactly once:
//   P[s][320][N] = X @ W1^T (K-split partials). A is K-blocked.
// Block: 8 waves; wave owns N-tile n0 = bx*128 + w*16; ALL 20 M-tiles kept in
// registers (160 acc VGPRs; __launch_bounds__ raises the VGPR cap so they
// stay in VGPRs), W fragment reused 20x per fetch.
// ---------------------------------------------------------------------------
#define BIGM 320
#define MT   20
__global__ void __launch_bounds__(256, 2)
k_gemm_big_splitk(const unsigned short* __restrict__ A,
                  const float* __restrict__ W,
                  float* __restrict__ P,
                  int N, int K, int Kc) {
  const int lane = threadIdx.x & 31;
  const int wave = threadIdx.x >> 5;
  const int n0 = blockIdx.x * 128 + wave * 16;
  const int lr = lane & 15;
  const int kh = lane >> 4;
  const int k0 = blockIdx.y * Kc;

  // single base pointer; all per-m offsets are immediates (m*1024 bytes)
  const unsigned short* Ab = A + (size_t)(k0 >> 5) * (BIGM * 32)
                               + (size_t)(lr * 32 + kh * 16);
  const float* Wp = W + (size_t)(n0 + lr) * K + (size_t)(kh * 16) + k0;

  v8f c[MT];
#pragma unroll
  for (int m = 0; m < MT; ++m) c[m] = {};

  const int iters = Kc >> 5;
#pragma unroll 1
  for (int it = 0; it < iters; ++it) {
    __builtin_prefetch((const void*)(Wp + 1024), 0, 1);
    v16bf bm = load_b_frag_f32(Wp);
#pragma unroll
    for (int m = 0; m < MT; ++m) {
      v16bf a = load_a_frag(Ab + m * 512);
      c[m] = __builtin_amdgcn_wmma_f32_16x16x32_bf16(false, a, false, bm,
                                                     (short)0, c[m], false, false);
    }
    Ab += BIGM * 32;
    Wp += 32;
  }

  float* Pb = P + (size_t)blockIdx.y * BIGM * N + (size_t)(kh * 8) * N
                + (size_t)(n0 + lr);
#pragma unroll
  for (int m = 0; m < MT; ++m) {
#pragma unroll
    for (int r = 0; r < 8; ++r) Pb[(size_t)(m * 16 + r) * N] = c[m][r];
  }
}

// D[i] = sum_s P[s][i]
__global__ void k_reduce_partials(const float* __restrict__ P,
                                  float* __restrict__ D, int n, int S) {
  int i = blockIdx.x * blockDim.x + threadIdx.x;
  if (i >= n) return;
  float acc = 0.f;
  for (int s = 0; s < S; ++s) acc += P[(size_t)s * n + i];
  D[i] = acc;
}

// ---------------------------------------------------------------------------
// Scan GEMM, bf16 weights (pre-converted): D[M][N] (+)= A[M][K] @ Wb[N][K]^T
// ---------------------------------------------------------------------------
template <int ACCUM>
__global__ void k_gemm_bf16(const unsigned short* __restrict__ A,
                            const unsigned short* __restrict__ Wb,
                            float* __restrict__ D, int M, int N, int K) {
  const int lane = threadIdx.x & 31;
  const int wave = threadIdx.x >> 5;
  const int m0 = blockIdx.y * 32;
  const int n0 = blockIdx.x * 128 + wave * 16;
  if (n0 >= N) return;                        // wave-uniform
  const bool twoM = (m0 + 16) < M;            // block-uniform
  const int lr = lane & 15;
  const int kh = lane >> 4;

  const unsigned short* Ap0 = A + (size_t)(m0 + lr) * K + (size_t)(kh * 16);
  const unsigned short* Ap1 = Ap0 + (size_t)16 * K;
  const unsigned short* Wp  = Wb + (size_t)(n0 + lr) * K + (size_t)(kh * 16);

  v8f c0 = {};
  v8f c1 = {};
  for (int k = 0; k < K; k += 32) {
    v16bf bm = load_w_frag_bf16(Wp + k);
    v16bf a0 = load_a_frag(Ap0 + k);
    c0 = __builtin_amdgcn_wmma_f32_16x16x32_bf16(false, a0, false, bm,
                                                 (short)0, c0, false, false);
    if (twoM) {
      v16bf a1 = load_a_frag(Ap1 + k);
      c1 = __builtin_amdgcn_wmma_f32_16x16x32_bf16(false, a1, false, bm,
                                                   (short)0, c1, false, false);
    }
  }
  const int cn = n0 + lr;
  const int cmb = m0 + kh * 8;
#pragma unroll
  for (int r = 0; r < 8; ++r) {
    size_t di = (size_t)(cmb + r) * N + cn;
    D[di] = ACCUM ? (D[di] + c0[r]) : c0[r];
  }
  if (twoM) {
#pragma unroll
    for (int r = 0; r < 8; ++r) {
      size_t di = (size_t)(cmb + 16 + r) * N + cn;
      D[di] = ACCUM ? (D[di] + c1[r]) : c1[r];
    }
  }
}

// Fallback GEMM, f32 weights converted in-registers (small-ws path)
template <int ACCUM>
__global__ void k_gemm_w32(const unsigned short* __restrict__ A,
                           const float* __restrict__ W,
                           float* __restrict__ D, int M, int N, int K) {
  const int lane = threadIdx.x & 31;
  const int wave = threadIdx.x >> 5;
  const int m0 = blockIdx.y * 32;
  const int n0 = blockIdx.x * 128 + wave * 16;
  if (n0 >= N) return;
  const bool twoM = (m0 + 16) < M;
  const int lr = lane & 15;
  const int kh = lane >> 4;

  const unsigned short* Ap0 = A + (size_t)(m0 + lr) * K + (size_t)(kh * 16);
  const unsigned short* Ap1 = Ap0 + (size_t)16 * K;
  const float*          Wp  = W + (size_t)(n0 + lr) * K + (size_t)(kh * 16);

  v8f c0 = {};
  v8f c1 = {};
  for (int k = 0; k < K; k += 32) {
    __builtin_prefetch((const void*)(Wp + k + 1024), 0, 1);
    v16bf bm = load_b_frag_f32(Wp + k);
    v16bf a0 = load_a_frag(Ap0 + k);
    c0 = __builtin_amdgcn_wmma_f32_16x16x32_bf16(false, a0, false, bm,
                                                 (short)0, c0, false, false);
    if (twoM) {
      v16bf a1 = load_a_frag(Ap1 + k);
      c1 = __builtin_amdgcn_wmma_f32_16x16x32_bf16(false, a1, false, bm,
                                                   (short)0, c1, false, false);
    }
  }
  const int cn = n0 + lr;
  const int cmb = m0 + kh * 8;
#pragma unroll
  for (int r = 0; r < 8; ++r) {
    size_t di = (size_t)(cmb + r) * N + cn;
    D[di] = ACCUM ? (D[di] + c0[r]) : c0[r];
  }
  if (twoM) {
#pragma unroll
    for (int r = 0; r < 8; ++r) {
      size_t di = (size_t)(cmb + 16 + r) * N + cn;
      D[di] = ACCUM ? (D[di] + c1[r]) : c1[r];
    }
  }
}

// ---------------------------------------------------------------------------
__global__ void k_snu_update(const float* __restrict__ drive,
                             float* __restrict__ s,
                             unsigned short* __restrict__ y, int n) {
  int i = blockIdx.x * blockDim.x + threadIdx.x;
  if (i >= n) return;
  float yp = (y[i] != 0) ? 1.f : 0.f;
  float v = drive[i] + L_TAU * s[i] * (1.f - yp);
  v = fmaxf(v, 0.f);
  s[i] = v;
  y[i] = (v > 0.f) ? (unsigned short)0x3F80u : (unsigned short)0u;
}

__global__ void k_out_update(const float* __restrict__ d_o,
                             float* __restrict__ s_o,
                             float* __restrict__ out, int t) {
  __shared__ float red[1024];
  int tid = threadIdx.x;
  int b = tid >> 6;
  int j = tid & 63;
  float v = fmaxf(d_o[tid] + L_TAU * s_o[tid], 0.f);
  s_o[tid] = v;
  red[tid] = v;
  __syncthreads();
#pragma unroll
  for (int off = 32; off > 0; off >>= 1) {
    if (j < off) red[tid] += red[tid + off];
    __syncthreads();
  }
  if (j == 0) out[b * 20 + t] = red[tid] * (1.f / 64.f);
}

// ---------------------------------------------------------------------------
extern "C" void kernel_launch(void* const* d_in, const int* in_sizes, int n_in,
                              void* d_out, int out_size, void* d_ws, size_t ws_size,
                              hipStream_t stream) {
  (void)in_sizes; (void)n_in; (void)out_size;
  const float* x  = (const float*)d_in[0];
  const float* W1 = (const float*)d_in[2];
  const float* V1 = (const float*)d_in[3];
  const float* W2 = (const float*)d_in[4];
  const float* V2 = (const float*)d_in[5];
  const float* W3 = (const float*)d_in[6];
  const float* V3 = (const float*)d_in[7];
  const float* W4 = (const float*)d_in[8];
  const float* V4 = (const float*)d_in[9];
  const float* Wo = (const float*)d_in[10];
  float* out = (float*)d_out;

  const int Bb = 16, T = 20, SPL = 16;
  const int D0 = 32768, D1 = 4096, D2 = 1024, D3 = 256, D4 = 64;

  char* ws = (char*)d_ws;
  size_t off = 0;
  auto carve = [&](size_t bytes) -> char* {
    char* p = ws + off;
    off += (bytes + 255) & ~(size_t)255;
    return p;
  };
  // ---- common region (~27 MB) ----
  unsigned short* Xb = (unsigned short*)carve((size_t)T * Bb * D0 * 2);
  float* drive1 = (float*)carve((size_t)T * Bb * D1 * 4);
  float* dbuf2  = (float*)carve((size_t)Bb * D2 * 4);
  float* dbuf3  = (float*)carve((size_t)Bb * D3 * 4);
  float* dbuf4  = (float*)carve((size_t)Bb * D4 * 4);
  float* dbufo  = (float*)carve((size_t)Bb * D4 * 4);
  char* state0 = ws + off;
  float* s1 = (float*)carve((size_t)Bb * D1 * 4);
  float* s2 = (float*)carve((size_t)Bb * D2 * 4);
  float* s3 = (float*)carve((size_t)Bb * D3 * 4);
  float* s4 = (float*)carve((size_t)Bb * D4 * 4);
  float* so = (float*)carve((size_t)Bb * D4 * 4);
  unsigned short* y1 = (unsigned short*)carve((size_t)Bb * D1 * 2);
  unsigned short* y2 = (unsigned short*)carve((size_t)Bb * D2 * 2);
  unsigned short* y3 = (unsigned short*)carve((size_t)Bb * D3 * 2);
  unsigned short* y4 = (unsigned short*)carve((size_t)Bb * D4 * 2);
  int state_dw = (int)(((ws + off) - state0) >> 2);

  // ---- optional full-speed region: split-K partials + bf16 weight cache ----
  float* Ppart = (float*)carve((size_t)SPL * T * Bb * D1 * 4);        // 84 MB
  unsigned short* V1b = (unsigned short*)carve((size_t)D1 * D1 * 2);  // 32 MB
  unsigned short* W2b = (unsigned short*)carve((size_t)D2 * D1 * 2);
  unsigned short* V2b = (unsigned short*)carve((size_t)D2 * D2 * 2);
  unsigned short* W3b = (unsigned short*)carve((size_t)D3 * D2 * 2);
  unsigned short* V3b = (unsigned short*)carve((size_t)D3 * D3 * 2);
  unsigned short* W4b = (unsigned short*)carve((size_t)D4 * D3 * 2);
  unsigned short* V4b = (unsigned short*)carve((size_t)D4 * D4 * 2);
  unsigned short* Wob = (unsigned short*)carve((size_t)D4 * D4 * 2);
  const bool full = (off <= ws_size);

  k_zero_u32<<<(state_dw + 255) / 256, 256, 0, stream>>>(
      (unsigned int*)state0, state_dw);

  int xtot = T * Bb * D0;
  if (full) {
    k_pack_x_kblk<<<(xtot + 255) / 256, 256, 0, stream>>>(x, Xb, xtot);
    // bf16 weight cache (read each f32 weight once)
    k_cvt_bf16<<<(D1 * D1 + 255) / 256, 256, 0, stream>>>(V1, V1b, D1 * D1);
    k_cvt_bf16<<<(D2 * D1 + 255) / 256, 256, 0, stream>>>(W2, W2b, D2 * D1);
    k_cvt_bf16<<<(D2 * D2 + 255) / 256, 256, 0, stream>>>(V2, V2b, D2 * D2);
    k_cvt_bf16<<<(D3 * D2 + 255) / 256, 256, 0, stream>>>(W3, W3b, D3 * D2);
    k_cvt_bf16<<<(D3 * D3 + 255) / 256, 256, 0, stream>>>(V3, V3b, D3 * D3);
    k_cvt_bf16<<<(D4 * D3 + 255) / 256, 256, 0, stream>>>(W4, W4b, D4 * D3);
    k_cvt_bf16<<<(D4 * D4 + 255) / 256, 256, 0, stream>>>(V4, V4b, D4 * D4);
    k_cvt_bf16<<<(D4 * D4 + 255) / 256, 256, 0, stream>>>(Wo, Wob, D4 * D4);
    // big GEMM: W1 streamed once; split-K partials then deterministic reduce
    k_gemm_big_splitk<<<dim3(D1 / 128, SPL), 256, 0, stream>>>(
        Xb, W1, Ppart, D1, D0, D0 / SPL);
    k_reduce_partials<<<(T * Bb * D1 + 255) / 256, 256, 0, stream>>>(
        Ppart, drive1, T * Bb * D1, SPL);

    for (int t = 0; t < T; ++t) {
      float* d1t = drive1 + (size_t)t * Bb * D1;
      k_gemm_bf16<1><<<dim3(D1 / 128, 1), 256, 0, stream>>>(y1, V1b, d1t, Bb, D1, D1);
      k_snu_update<<<(Bb * D1 + 255) / 256, 256, 0, stream>>>(d1t, s1, y1, Bb * D1);
      k_gemm_bf16<0><<<dim3(D2 / 128, 1), 256, 0, stream>>>(y1, W2b, dbuf2, Bb, D2, D1);
      k_gemm_bf16<1><<<dim3(D2 / 128, 1), 256, 0, stream>>>(y2, V2b, dbuf2, Bb, D2, D2);
      k_snu_update<<<(Bb * D2 + 255) / 256, 256, 0, stream>>>(dbuf2, s2, y2, Bb * D2);
      k_gemm_bf16<0><<<dim3(2, 1), 256, 0, stream>>>(y2, W3b, dbuf3, Bb, D3, D2);
      k_gemm_bf16<1><<<dim3(2, 1), 256, 0, stream>>>(y3, V3b, dbuf3, Bb, D3, D3);
      k_snu_update<<<(Bb * D3 + 255) / 256, 256, 0, stream>>>(dbuf3, s3, y3, Bb * D3);
      k_gemm_bf16<0><<<dim3(1, 1), 256, 0, stream>>>(y3, W4b, dbuf4, Bb, D4, D3);
      k_gemm_bf16<1><<<dim3(1, 1), 256, 0, stream>>>(y4, V4b, dbuf4, Bb, D4, D4);
      k_snu_update<<<(Bb * D4 + 255) / 256, 256, 0, stream>>>(dbuf4, s4, y4, Bb * D4);
      k_gemm_bf16<0><<<dim3(1, 1), 256, 0, stream>>>(y4, Wob, dbufo, Bb, D4, D4);
      k_out_update<<<1, 1024, 0, stream>>>(dbufo, so, out, t);
    }
  } else {
    // small-workspace fallback: on-the-fly f32->bf16 conversion everywhere
    k_pack_x_bf16<<<(xtot + 255) / 256, 256, 0, stream>>>(x, Xb, xtot);
    k_gemm_w32<0><<<dim3(D1 / 128, (T * Bb) / 32), 256, 0, stream>>>(
        Xb, W1, drive1, T * Bb, D1, D0);
    for (int t = 0; t < T; ++t) {
      float* d1t = drive1 + (size_t)t * Bb * D1;
      k_gemm_w32<1><<<dim3(D1 / 128, 1), 256, 0, stream>>>(y1, V1, d1t, Bb, D1, D1);
      k_snu_update<<<(Bb * D1 + 255) / 256, 256, 0, stream>>>(d1t, s1, y1, Bb * D1);
      k_gemm_w32<0><<<dim3(D2 / 128, 1), 256, 0, stream>>>(y1, W2, dbuf2, Bb, D2, D1);
      k_gemm_w32<1><<<dim3(D2 / 128, 1), 256, 0, stream>>>(y2, V2, dbuf2, Bb, D2, D2);
      k_snu_update<<<(Bb * D2 + 255) / 256, 256, 0, stream>>>(dbuf2, s2, y2, Bb * D2);
      k_gemm_w32<0><<<dim3(2, 1), 256, 0, stream>>>(y2, W3, dbuf3, Bb, D3, D2);
      k_gemm_w32<1><<<dim3(2, 1), 256, 0, stream>>>(y3, V3, dbuf3, Bb, D3, D3);
      k_snu_update<<<(Bb * D3 + 255) / 256, 256, 0, stream>>>(dbuf3, s3, y3, Bb * D3);
      k_gemm_w32<0><<<dim3(1, 1), 256, 0, stream>>>(y3, W4, dbuf4, Bb, D4, D3);
      k_gemm_w32<1><<<dim3(1, 1), 256, 0, stream>>>(y4, V4, dbuf4, Bb, D4, D4);
      k_snu_update<<<(Bb * D4 + 255) / 256, 256, 0, stream>>>(dbuf4, s4, y4, Bb * D4);
      k_gemm_w32<0><<<dim3(1, 1), 256, 0, stream>>>(y4, Wo, dbufo, Bb, D4, D4);
      k_out_update<<<1, 1024, 0, stream>>>(dbufo, so, out, t);
    }
  }
}
